// TransformerBlock_35433480192374
// MI455X (gfx1250) — compile-verified, matrix operand order checked by reading, not run on hardware
//
#include <hip/hip_runtime.h>
#include <hip/hip_bf16.h>

// ---------------------------------------------------------------------------
// TransformerBlock for MI455X (gfx1250, wave32, WMMA).
//
// Exploits the reference's einsum semantics: softmax rows sum to 1 and the
// 'k'/'v' einsum indices are independent, so Q/K drop out and the attention
// output reduces to  Block[n] = (x @ wv_sum) reinterpreted (256,1024), where
// wv_sum folds the 16 head-columns of w_v.  Heavy work is the FFN, run as
// bf16 WMMA GEMMs with fp32 accumulation.
// ---------------------------------------------------------------------------

typedef __attribute__((ext_vector_type(16))) __bf16 v16bf;
typedef __attribute__((ext_vector_type(8)))  __bf16 bf16x8;
typedef __attribute__((ext_vector_type(8)))  float  v8f;
typedef __attribute__((ext_vector_type(4)))  float  f32x4;

static constexpr int BATCH = 4, SEQ = 4096, EMB = 1024, HIDDEN = 4096;
static constexpr int NHEADS = 16, HDIM = 64;
static constexpr float LN_EPS = 1e-5f;

// ---------------------------------------------------------------------------
// Fold the 16 head-slices of w_v and emit transposed bf16: wvT[d][k] (64x1024)
// ---------------------------------------------------------------------------
__global__ void fold_wv_kernel(const float* __restrict__ w_v,
                               __bf16* __restrict__ wvT) {
  int t = blockIdx.x * blockDim.x + threadIdx.x;   // 64*1024 threads
  int d = t >> 10;
  int k = t & 1023;
  float s = 0.f;
#pragma unroll
  for (int h = 0; h < NHEADS; ++h) s += w_v[(size_t)k * EMB + h * HDIM + d];
  wvT[(size_t)d * EMB + k] = (__bf16)s;
}

// ---------------------------------------------------------------------------
// Transpose fp32 (K x N) -> bf16 (N x K) via LDS tiles (coalesced both ways).
// ---------------------------------------------------------------------------
__global__ void transpose_to_bf16_kernel(const float* __restrict__ in,
                                         __bf16* __restrict__ out,
                                         int K, int N) {
  __shared__ float tile[32][33];
  const int n0 = blockIdx.x * 32, k0 = blockIdx.y * 32;
  const int tx = threadIdx.x, ty = threadIdx.y;   // (32, 8)
#pragma unroll
  for (int i = 0; i < 4; ++i)
    tile[ty + 8 * i][tx] = in[(size_t)(k0 + ty + 8 * i) * N + (n0 + tx)];
  __syncthreads();
#pragma unroll
  for (int i = 0; i < 4; ++i)
    out[(size_t)(n0 + ty + 8 * i) * K + (k0 + tx)] = (__bf16)tile[tx][ty + 8 * i];
}

// ---------------------------------------------------------------------------
// bf16 WMMA GEMM:  C(MxN) = A(MxK) * Bt(NxK)^T   (Bt pre-transposed bf16)
//   A_F32: A is fp32 (converted to bf16 while staging), else A is bf16.
//   EPI:   0 = fp32 store, 1 = fp32 + bias, 2 = bf16 + bias + relu, 3 = bf16.
// 256 threads = 8 waves; double-buffered LDS; KSTEP=64 (2 wmma k-chunks).
// ---------------------------------------------------------------------------
template <int BM, int BN, int WAVES_M, int WAVES_N, bool A_F32, int EPI>
__global__ __launch_bounds__(256) void gemm_wmma_kernel(
    const void* __restrict__ Aptr, const __bf16* __restrict__ Bt,
    const float* __restrict__ bias, void* __restrict__ Cptr,
    int M, int N, int K) {
  static_assert(WAVES_M * WAVES_N == 8, "8 waves per block");
  constexpr int KSTEP = 64;
  constexpr int KPAD  = 72;                 // 144B rows: 16B aligned, bank-skewed
  constexpr int MT = BM / WAVES_M / 16;     // 16x16 tiles per wave (M)
  constexpr int NT = BN / WAVES_N / 16;     // 16x16 tiles per wave (N)

  __shared__ __align__(16) __bf16 sA[2][BM * KPAD];
  __shared__ __align__(16) __bf16 sB[2][BN * KPAD];

  const int tid  = threadIdx.x;
  const int lane = tid & 31;
  const int wave = tid >> 5;
  const int wm = (wave % WAVES_M) * (MT * 16);
  const int wn = (wave / WAVES_M) * (NT * 16);
  const int bm = blockIdx.x * BM;
  const int bn = blockIdx.y * BN;

  // staging geometry
  constexpr int A_CHUNK = BM * KSTEP / 256;       // elems per thread
  constexpr int A_TPR   = KSTEP / A_CHUNK;
  constexpr int B_CHUNK = BN * KSTEP / 256;
  constexpr int B_TPR   = KSTEP / B_CHUNK;
  const int a_row = tid / A_TPR, a_k = (tid % A_TPR) * A_CHUNK;
  const int b_row = tid / B_TPR, b_k = (tid % B_TPR) * B_CHUNK;

  const float*  Af = (const float*)Aptr;
  const __bf16* Ab = (const __bf16*)Aptr;

  v8f acc[MT][NT];
#pragma unroll
  for (int i = 0; i < MT; ++i)
#pragma unroll
    for (int j = 0; j < NT; ++j)
#pragma unroll
      for (int r = 0; r < 8; ++r) acc[i][j][r] = 0.f;

  // register staging (overlap global latency with WMMA)
  f32x4  rAf[A_CHUNK / 4];
  bf16x8 rAb[A_CHUNK / 8];
  bf16x8 rB [B_CHUNK / 8];

  auto load_regs = [&](int kbase) {
    if (A_F32) {
      const float* p = Af + (size_t)(bm + a_row) * K + kbase + a_k;
#pragma unroll
      for (int i = 0; i < A_CHUNK / 4; ++i) rAf[i] = *(const f32x4*)(p + i * 4);
    } else {
      const __bf16* p = Ab + (size_t)(bm + a_row) * K + kbase + a_k;
#pragma unroll
      for (int i = 0; i < A_CHUNK / 8; ++i) rAb[i] = *(const bf16x8*)(p + i * 8);
    }
    const __bf16* q = Bt + (size_t)(bn + b_row) * K + kbase + b_k;
#pragma unroll
    for (int i = 0; i < B_CHUNK / 8; ++i) rB[i] = *(const bf16x8*)(q + i * 8);
  };

  auto store_lds = [&](int buf) {
    __bf16* a = &sA[buf][a_row * KPAD + a_k];
    if (A_F32) {
      const float* fa = (const float*)rAf;
#pragma unroll
      for (int i = 0; i < A_CHUNK / 8; ++i) {
        bf16x8 v;
#pragma unroll
        for (int j = 0; j < 8; ++j) v[j] = (__bf16)fa[i * 8 + j];
        *(bf16x8*)(a + i * 8) = v;
      }
    } else {
#pragma unroll
      for (int i = 0; i < A_CHUNK / 8; ++i) *(bf16x8*)(a + i * 8) = rAb[i];
    }
    __bf16* b = &sB[buf][b_row * KPAD + b_k];
#pragma unroll
    for (int i = 0; i < B_CHUNK / 8; ++i) *(bf16x8*)(b + i * 8) = rB[i];
  };

  // ISA A/B fragment layout (16-bit, 16x32): lane m -> row m, K{0..7,16..23};
  // lane m+16 -> row m, K{8..15,24..31}.
  auto frag = [&](const __bf16* base, int row0, int k0) -> v16bf {
    const __bf16* p = base + (row0 + (lane & 15)) * KPAD + k0 + ((lane >> 4) << 3);
    bf16x8 lo = *(const bf16x8*)p;
    bf16x8 hi = *(const bf16x8*)(p + 16);
    return __builtin_shufflevector(lo, hi, 0, 1, 2, 3, 4, 5, 6, 7,
                                   8, 9, 10, 11, 12, 13, 14, 15);
  };

  auto compute = [&](int buf) {
#pragma unroll
    for (int k0 = 0; k0 < KSTEP; k0 += 32) {
      v16bf afr[MT], bfr[NT];
#pragma unroll
      for (int i = 0; i < MT; ++i) afr[i] = frag(sA[buf], wm + i * 16, k0);
#pragma unroll
      for (int j = 0; j < NT; ++j) bfr[j] = frag(sB[buf], wn + j * 16, k0);
#pragma unroll
      for (int i = 0; i < MT; ++i)
#pragma unroll
        for (int j = 0; j < NT; ++j)
          acc[i][j] = __builtin_amdgcn_wmma_f32_16x16x32_bf16(
              false, afr[i], false, bfr[j], (short)0, acc[i][j], false, false);
    }
  };

  const int nk = K / KSTEP;
  load_regs(0);
  store_lds(0);
  __syncthreads();
  for (int kt = 0; kt < nk; ++kt) {
    const int cur = kt & 1;
    if (kt + 1 < nk) load_regs((kt + 1) * KSTEP);
    compute(cur);
    if (kt + 1 < nk) store_lds(cur ^ 1);
    __syncthreads();
  }

  // Epilogue. C/D layout: VGPR j, lanes 0-15 -> (M=j, N=lane);
  // lanes 16-31 -> (M=8+j, N=lane-16).
  float*  Cf = (float*)Cptr;
  __bf16* Cb = (__bf16*)Cptr;
  const int rhalf = (lane >> 4) * 8;
  const int cin   = lane & 15;
#pragma unroll
  for (int i = 0; i < MT; ++i) {
#pragma unroll
    for (int j = 0; j < NT; ++j) {
      const int col = bn + wn + j * 16 + cin;
      const float bv = (EPI == 1 || EPI == 2) ? bias[col] : 0.f;
#pragma unroll
      for (int r = 0; r < 8; ++r) {
        const int row = bm + wm + i * 16 + rhalf + r;
        float v = acc[i][j][r] + bv;
        if (EPI == 2) v = v > 0.f ? v : 0.f;
        const size_t off = (size_t)row * N + col;
        if (EPI == 0 || EPI == 1) Cf[off] = v;
        else                      Cb[off] = (__bf16)v;
      }
    }
  }
}

// ---------------------------------------------------------------------------
// Residual + LayerNorm, one 256-thread block per row of 1024.
// attn_out[n,l,:] == ab[n*256 + (l%256), :]  (period-256 repetition).
// ---------------------------------------------------------------------------
__global__ __launch_bounds__(256) void ln1_kernel(
    const float* __restrict__ x, const float* __restrict__ ab,
    const float* __restrict__ w, const float* __restrict__ b,
    float* __restrict__ h1) {
  __shared__ float red[256];
  const int row = blockIdx.x;                 // 0..BATCH*SEQ-1
  const int n = row >> 12, l = row & 4095;
  const float* xr = x + (size_t)row * EMB;
  const float* ar = ab + (size_t)(n * 256 + (l & 255)) * EMB;
  const int t = threadIdx.x;
  float v[4];
  float s = 0.f;
#pragma unroll
  for (int i = 0; i < 4; ++i) { v[i] = xr[i * 256 + t] + ar[i * 256 + t]; s += v[i]; }
  red[t] = s; __syncthreads();
  for (int o = 128; o > 0; o >>= 1) { if (t < o) red[t] += red[t + o]; __syncthreads(); }
  const float mu = red[0] * (1.f / EMB);
  __syncthreads();
  float vs = 0.f;
#pragma unroll
  for (int i = 0; i < 4; ++i) { float d = v[i] - mu; vs += d * d; }
  red[t] = vs; __syncthreads();
  for (int o = 128; o > 0; o >>= 1) { if (t < o) red[t] += red[t + o]; __syncthreads(); }
  const float rstd = rsqrtf(red[0] * (1.f / EMB) + LN_EPS);
  float* hr = h1 + (size_t)row * EMB;
#pragma unroll
  for (int i = 0; i < 4; ++i) {
    const int e = i * 256 + t;
    hr[e] = (v[i] - mu) * rstd * w[e] + b[e];
  }
}

// In-place: y = LN(y + h1) over each row of d_out.
__global__ __launch_bounds__(256) void ln2_kernel(
    const float* __restrict__ h1, const float* __restrict__ w,
    const float* __restrict__ b, float* __restrict__ y) {
  __shared__ float red[256];
  const int row = blockIdx.x;
  const float* hr = h1 + (size_t)row * EMB;
  float* yr = y + (size_t)row * EMB;
  const int t = threadIdx.x;
  float v[4];
  float s = 0.f;
#pragma unroll
  for (int i = 0; i < 4; ++i) { v[i] = yr[i * 256 + t] + hr[i * 256 + t]; s += v[i]; }
  red[t] = s; __syncthreads();
  for (int o = 128; o > 0; o >>= 1) { if (t < o) red[t] += red[t + o]; __syncthreads(); }
  const float mu = red[0] * (1.f / EMB);
  __syncthreads();
  float vs = 0.f;
#pragma unroll
  for (int i = 0; i < 4; ++i) { float d = v[i] - mu; vs += d * d; }
  red[t] = vs; __syncthreads();
  for (int o = 128; o > 0; o >>= 1) { if (t < o) red[t] += red[t + o]; __syncthreads(); }
  const float rstd = rsqrtf(red[0] * (1.f / EMB) + LN_EPS);
#pragma unroll
  for (int i = 0; i < 4; ++i) {
    const int e = i * 256 + t;
    yr[e] = (v[i] - mu) * rstd * w[e] + b[e];
  }
}

// ---------------------------------------------------------------------------
extern "C" void kernel_launch(void* const* d_in, const int* in_sizes, int n_in,
                              void* d_out, int out_size, void* d_ws, size_t ws_size,
                              hipStream_t stream) {
  const float* x     = (const float*)d_in[0];
  // d_in[1]=w_q, d_in[2]=w_k: mathematically cancel (softmax rows sum to 1,
  // and 'k'/'v' are independent einsum sum indices) -> not needed.
  const float* w_v   = (const float*)d_in[3];
  const float* w_o   = (const float*)d_in[4];
  const float* b_o   = (const float*)d_in[5];
  const float* ln1_w = (const float*)d_in[6];
  const float* ln1_b = (const float*)d_in[7];
  const float* ln2_w = (const float*)d_in[8];
  const float* ln2_b = (const float*)d_in[9];
  const float* w_ff1 = (const float*)d_in[10];
  const float* b_ff1 = (const float*)d_in[11];
  const float* w_ff2 = (const float*)d_in[12];
  const float* b_ff2 = (const float*)d_in[13];

  char* ws = (char*)d_ws;
  auto alloc = [&](size_t bytes) {
    char* p = ws;
    ws += (bytes + 255) & ~(size_t)255;
    return p;
  };
  __bf16* wvT   = (__bf16*)alloc((size_t)HDIM * EMB * 2);           // 64x1024
  __bf16* woT   = (__bf16*)alloc((size_t)EMB * EMB * 2);            // 1024x1024
  __bf16* wff1T = (__bf16*)alloc((size_t)HIDDEN * EMB * 2);         // 4096x1024
  __bf16* wff2T = (__bf16*)alloc((size_t)EMB * HIDDEN * 2);         // 1024x4096
  __bf16* vsum  = (__bf16*)alloc((size_t)BATCH * SEQ * HDIM * 2);   // 16384x64
  float*  abuf  = (float*)alloc((size_t)BATCH * 256 * EMB * 4);     // 1024x1024
  float*  h1    = (float*)alloc((size_t)BATCH * SEQ * EMB * 4);     // 16384x1024
  __bf16* ffh   = (__bf16*)alloc((size_t)BATCH * SEQ * HIDDEN * 2); // 16384x4096

  const int M = BATCH * SEQ;   // 16384
  const dim3 tb(32, 8);

  // Weight prep (bf16, transposed to NxK)
  fold_wv_kernel<<<(HDIM * EMB) / 256, 256, 0, stream>>>(w_v, wvT);
  transpose_to_bf16_kernel<<<dim3(EMB / 32, EMB / 32), tb, 0, stream>>>(w_o, woT, EMB, EMB);
  transpose_to_bf16_kernel<<<dim3(HIDDEN / 32, EMB / 32), tb, 0, stream>>>(w_ff1, wff1T, EMB, HIDDEN);
  transpose_to_bf16_kernel<<<dim3(EMB / 32, HIDDEN / 32), tb, 0, stream>>>(w_ff2, wff2T, HIDDEN, EMB);

  // Vsum = x @ wv_sum  (16384x64, bf16 out)
  gemm_wmma_kernel<128, 64, 8, 1, true, 3>
      <<<dim3(M / 128, 1), 256, 0, stream>>>(x, wvT, nullptr, vsum, M, HDIM, EMB);

  // attn_out blocks = Vsum(as 1024x1024 bf16) @ w_o + b_o  (fp32 out)
  gemm_wmma_kernel<128, 128, 4, 2, false, 1>
      <<<dim3((BATCH * 256) / 128, EMB / 128), 256, 0, stream>>>(
          vsum, woT, b_o, abuf, BATCH * 256, EMB, EMB);

  // h1 = LN(attn_out + x)
  ln1_kernel<<<M, 256, 0, stream>>>(x, abuf, ln1_w, ln1_b, h1);

  // ffh = relu(h1 @ w_ff1 + b_ff1)  (bf16 out)
  gemm_wmma_kernel<128, 128, 4, 2, true, 2>
      <<<dim3(M / 128, HIDDEN / 128), 256, 0, stream>>>(
          h1, wff1T, b_ff1, ffh, M, HIDDEN, EMB);

  // d_out = ffh @ w_ff2 + b_ff2  (fp32)
  gemm_wmma_kernel<128, 128, 4, 2, false, 1>
      <<<dim3(M / 128, EMB / 128), 256, 0, stream>>>(
          ffh, wff2T, b_ff2, d_out, M, EMB, HIDDEN);

  // d_out = LN(d_out + h1)  (in place)
  ln2_kernel<<<M, 256, 0, stream>>>(h1, ln2_w, ln2_b, (float*)d_out);
}
